// RNN_39737037422829
// MI455X (gfx1250) — matrix operand order repaired and checked by reading, not compiled, not run on hardware
//
#include <hip/hip_runtime.h>
#include <hip/hip_bf16.h>
#include <math.h>

typedef __bf16 bf16_t;
typedef __bf16 v16bf __attribute__((ext_vector_type(16)));
typedef __bf16 v8bf  __attribute__((ext_vector_type(8)));
typedef float  v8f   __attribute__((ext_vector_type(8)));

static constexpr int B = 64, S = 512, I = 512, H = 512;
static constexpr int K = I + H;           // 1024: concat [x_t | h_{t-1}]
static constexpr int NT = H / 16;         // 32 N-tiles -> 32 workgroups
static constexpr int LDS_STRIDE = K + 16; // pad to reduce LDS bank conflicts

// Fast tanh: native V_TANH_F32 on gfx1250 if the builtin exists.
__device__ __forceinline__ float fast_tanh(float x) {
#if __has_builtin(__builtin_amdgcn_tanhf)
    return __builtin_amdgcn_tanhf(x);
#else
    return tanhf(x);
#endif
}

// ---------------- helper kernels ----------------

__global__ void k_f32_to_bf16(const float* __restrict__ in, bf16_t* __restrict__ out, int n) {
    int i = blockIdx.x * blockDim.x + threadIdx.x;
    if (i < n) out[i] = (bf16_t)in[i];
}

// Wcat[n][k] = k < I ? W_ih[n][k] : W_hh[n][k-I]   (bf16, row-major H x K)
__global__ void k_build_wcat(const float* __restrict__ Wih, const float* __restrict__ Whh,
                             bf16_t* __restrict__ Wcat) {
    int i = blockIdx.x * blockDim.x + threadIdx.x;
    if (i >= H * K) return;
    int n = i >> 10;          // K == 1024
    int k = i & (K - 1);
    float v = (k < I) ? Wih[n * I + k] : Whh[n * H + (k - I)];
    Wcat[i] = (bf16_t)v;
}

__global__ void k_zero_u32(unsigned* __restrict__ p, int n) {
    int i = blockIdx.x * blockDim.x + threadIdx.x;
    if (i < n) p[i] = 0u;
}

// x-part partial accumulator: 16 WMMAs over K=0..511 (input slice, no h dependence)
__device__ __forceinline__ v8f xpart_acc(const bf16_t* __restrict__ inrow,
                                         const bf16_t* __restrict__ Wl,
                                         int rlane, int khalf) {
    v8f acc = {0.f, 0.f, 0.f, 0.f, 0.f, 0.f, 0.f, 0.f};
#pragma unroll
    for (int kt = 0; kt < I / 32; ++kt) {
        const int kb = kt * 32 + khalf * 16;
        v8bf lo = *(const v8bf*)(inrow + kb);
        v8bf hi = *(const v8bf*)(inrow + kb + 8);
        v16bf a = __builtin_shufflevector(lo, hi,
                      0, 1, 2, 3, 4, 5, 6, 7, 8, 9, 10, 11, 12, 13, 14, 15);
        v16bf b = *(const v16bf*)(&Wl[rlane * LDS_STRIDE + kt * 32 + khalf * 16]);
        acc = __builtin_amdgcn_wmma_f32_16x16x32_bf16(
                  false, a, false, b, (short)0, acc, false, false);
    }
    return acc;
}

// ---------------- persistent scan kernel (one RNN layer) ----------------
//
// 32 workgroups (one per 16-col N-tile) x 4 waves (one per 16-row M-tile).
// Per step the *dependent* work is only: load h_{t-1} tile (L2), 16 WMMAs
// against register-resident W_hh tiles, native-tanh epilogue, barrier. The
// input-half GEMM for step t+1 is computed between barrier-signal and
// barrier-wait, hiding it behind the grid synchronization.

__launch_bounds__(128, 1)
__global__ void k_rnn_scan(const bf16_t* __restrict__ inbf,  // (B,S,512) layer input, bf16
                           const bf16_t* __restrict__ Wcat,  // (H,K) fused weights, bf16
                           const float*  __restrict__ bias,  // (H)
                           bf16_t* __restrict__ hA,          // (B,H) state ping (holds h_{-1})
                           bf16_t* __restrict__ hB,          // (B,H) state pong
                           bf16_t* __restrict__ out_bf,      // (B,S,H) bf16 h_t, or null
                           float*  __restrict__ y,           // (B,S,H) f32 h_t, or null
                           float*  __restrict__ hfinal,      // (B,H) final state, f32
                           unsigned* __restrict__ counter)   // grid-barrier counter (pre-zeroed)
{
    const int lane  = threadIdx.x & 31;
    const int wave  = threadIdx.x >> 5;   // 0..3  -> M-tile
    const int nt    = blockIdx.x;         // 0..31 -> N-tile
    const int rlane = lane & 15;
    const int khalf = lane >> 4;          // 0 | 1 (lower/upper lane half)
    const int mrow0 = wave * 16;
    const int nglob = nt * 16 + rlane;
    const float bv  = bias[nglob];

    // Stage this workgroup's 16 x 1024 weight slab into LDS once.
    __shared__ __align__(16) bf16_t Wl[16 * LDS_STRIDE];
    for (int i = threadIdx.x; i < 16 * K; i += 128) {
        int r = i >> 10;
        int k = i & (K - 1);
        Wl[r * LDS_STRIDE + k] = Wcat[(size_t)(nt * 16 + r) * K + k];
    }
    __syncthreads();

    // Pin the recurrent-half (W_hh) B-tiles in registers: 16 x v16bf = 128 VGPRs.
    v16bf bh[H / 32];
#pragma unroll
    for (int kt = 0; kt < H / 32; ++kt) {
        bh[kt] = *(const v16bf*)(
            &Wl[rlane * LDS_STRIDE + (I / 32 + kt) * 32 + khalf * 16]);
    }

    bf16_t* hr = hA;   // read h_{t-1}
    bf16_t* hw = hB;   // write h_t

    // Prime the pipeline: x-part for t = 0.
    v8f accx = xpart_acc(inbf + ((size_t)(mrow0 + rlane) * S + 0) * I,
                         Wl, rlane, khalf);

    for (int t = 0; t < S; ++t) {
        // ---- dependent half: h_{t-1} @ W_hh^T on top of accx ----
        v8f acc = accx;
        const bf16_t* hrow = hr + (size_t)(mrow0 + rlane) * H;
#pragma unroll
        for (int kt = 0; kt < H / 32; ++kt) {
            const int kb = kt * 32 + khalf * 16;
            v8bf lo = *(const v8bf*)(hrow + kb);
            v8bf hi = *(const v8bf*)(hrow + kb + 8);
            v16bf a = __builtin_shufflevector(lo, hi,
                          0, 1, 2, 3, 4, 5, 6, 7, 8, 9, 10, 11, 12, 13, 14, 15);
            acc = __builtin_amdgcn_wmma_f32_16x16x32_bf16(
                      false, a, false, bh[kt], (short)0, acc, false, false);
        }

        // ---- epilogue: C/D layout -> VGPR j is row (mrow0 + khalf*8 + j) ----
#pragma unroll
        for (int j = 0; j < 8; ++j) {
            const int row = mrow0 + khalf * 8 + j;
            float v = fast_tanh(acc[j] + bv);
            hw[(size_t)row * H + nglob] = (bf16_t)v;
            if (out_bf) out_bf[((size_t)row * S + t) * H + nglob] = (bf16_t)v;
            if (y)      y[((size_t)row * S + t) * H + nglob]      = v;
            if (t == S - 1) hfinal[(size_t)row * H + nglob] = v;
        }

        // ---- release: make h_t visible, signal arrival ----
        __threadfence();
        __syncthreads();
        if (threadIdx.x == 0) atomicAdd(counter, 1u);

        // ---- overlap: x-part GEMM for t+1 while other WGs catch up ----
        if (t + 1 < S) {
            accx = xpart_acc(inbf + ((size_t)(mrow0 + rlane) * S + (t + 1)) * I,
                             Wl, rlane, khalf);
        }

        // ---- acquire: wait for all WGs' h_t, invalidate stale lines ----
        if (threadIdx.x == 0) {
            const unsigned want = (unsigned)gridDim.x * (unsigned)(t + 1);
            while (atomicAdd(counter, 0u) < want) {
                __builtin_amdgcn_s_sleep(2);
            }
        }
        __syncthreads();
        __threadfence();

        bf16_t* tmp = hr; hr = hw; hw = tmp;   // ping-pong state
    }
}

// ---------------- host launcher ----------------

extern "C" void kernel_launch(void* const* d_in, const int* in_sizes, int n_in,
                              void* d_out, int out_size, void* d_ws, size_t ws_size,
                              hipStream_t stream) {
    (void)in_sizes; (void)n_in; (void)out_size;

    const float* x    = (const float*)d_in[0];   // (B,S,I)
    const float* h0   = (const float*)d_in[1];   // (2,B,H)
    const float* Wih0 = (const float*)d_in[2];   // (H,I)
    const float* b0   = (const float*)d_in[3];   // (H)
    const float* Whh0 = (const float*)d_in[4];   // (H,H)
    const float* Wih1 = (const float*)d_in[5];   // (H,H)
    const float* b1   = (const float*)d_in[6];   // (H)
    const float* Whh1 = (const float*)d_in[7];   // (H,H)
    float* out = (float*)d_out;                  // (B,S,H) output ++ (2,B,H) h_n

    // Workspace carve-up (256B aligned bumps).
    char* ws = (char*)d_ws;
    size_t off = 0;
    auto bump = [&](size_t bytes) -> char* {
        char* p = ws + off;
        off = (off + bytes + 255) & ~(size_t)255;
        return p;
    };
    bf16_t*   xbf   = (bf16_t*)bump((size_t)B * S * I * sizeof(bf16_t)); // 32 MB
    bf16_t*   h0all = (bf16_t*)bump((size_t)B * S * H * sizeof(bf16_t)); // 32 MB
    bf16_t*   wc0   = (bf16_t*)bump((size_t)H * K * sizeof(bf16_t));     //  1 MB
    bf16_t*   wc1   = (bf16_t*)bump((size_t)H * K * sizeof(bf16_t));     //  1 MB
    bf16_t*   hA0   = (bf16_t*)bump((size_t)B * H * sizeof(bf16_t));
    bf16_t*   hB0   = (bf16_t*)bump((size_t)B * H * sizeof(bf16_t));
    bf16_t*   hA1   = (bf16_t*)bump((size_t)B * H * sizeof(bf16_t));
    bf16_t*   hB1   = (bf16_t*)bump((size_t)B * H * sizeof(bf16_t));
    unsigned* ctr   = (unsigned*)bump(2 * sizeof(unsigned));
    if (off > ws_size) return;  // insufficient scratch

    // Phase 0: barriers to zero, inputs/weights/state to bf16 (all parallel).
    k_zero_u32<<<1, 32, 0, stream>>>(ctr, 2);
    {
        int n = B * S * I;
        k_f32_to_bf16<<<(n + 255) / 256, 256, 0, stream>>>(x, xbf, n);
    }
    {
        int n = B * H;
        k_f32_to_bf16<<<(n + 255) / 256, 256, 0, stream>>>(h0, hA0, n);
        k_f32_to_bf16<<<(n + 255) / 256, 256, 0, stream>>>(h0 + (size_t)B * H, hA1, n);
    }
    {
        int n = H * K;
        k_build_wcat<<<(n + 255) / 256, 256, 0, stream>>>(Wih0, Whh0, wc0);
        k_build_wcat<<<(n + 255) / 256, 256, 0, stream>>>(Wih1, Whh1, wc1);
    }

    float* y  = out;                       // (B,S,H)
    float* hn = out + (size_t)B * S * H;   // (2,B,H)

    // Layer 0: consume x, produce bf16 h0_all + final state hn[0].
    k_rnn_scan<<<NT, 128, 0, stream>>>(xbf, wc0, b0, hA0, hB0,
                                       h0all, nullptr, hn, ctr + 0);
    // Layer 1: consume h0_all, produce f32 y + final state hn[1].
    k_rnn_scan<<<NT, 128, 0, stream>>>(h0all, wc1, b1, hA1, hB1,
                                       nullptr, y, hn + (size_t)B * H, ctr + 1);
}
